// GraphTransformerEncoderLayer_20091857010828
// MI455X (gfx1250) — compile-verified, matrix operand order checked by reading, not run on hardware
//
#include <hip/hip_runtime.h>
#include <hip/hip_bf16.h>
#include <math.h>

typedef __attribute__((ext_vector_type(16))) _Float16 v16h;
typedef __attribute__((ext_vector_type(8)))  _Float16 h8;
typedef __attribute__((ext_vector_type(8)))  float    v8f;
typedef __attribute__((ext_vector_type(4)))  int      v4i;
typedef __attribute__((ext_vector_type(8)))  int      v8i;

#define D_MODEL 512
#define D_FF    2048
#define NHEAD   8
#define DHEAD   64
#define NBATCH  64
#define TRIPLE  1536
#define NEGBIG  -1.0e9f

union AFrag {
  v16h v;
  h8   h[2];
  _Float16 e[16];
};

__device__ __forceinline__ v8f wmma_f16(const v16h& a, const v16h& b, const v8f& c) {
  // D = A(16x32 f16) * B(32x16 f16) + C(16x16 f32)
  return __builtin_amdgcn_wmma_f32_16x16x32_f16(false, a, false, b, (short)0, c, false, false);
}

// ---------------------------------------------------------------------------
// TDM: issue a 2D tensor_load_to_lds of a [256 rows x 64 halves] weight tile
// (row stride = K halves) into LDS at lds_off. D# built per CDNA5 ISA:
// group0: count=1|lds_addr|global_addr|type=2 ; group1: data_size=2B, dims.
// Wave-level op (EXEC ignored); caller must restrict to one wave per block.
// ---------------------------------------------------------------------------
__device__ __forceinline__ void tdm_load_wtile(const _Float16* gptr, unsigned lds_off,
                                               int K, int Nout) {
  unsigned long long ga = (unsigned long long)gptr;
  v4i g0;
  g0[0] = 1;                                                   // count=1 (valid user D#)
  g0[1] = (int)lds_off;                                        // lds_addr (bytes)
  g0[2] = (int)(ga & 0xffffffffull);                           // global_addr[31:0]
  g0[3] = (int)(((ga >> 32) & 0x01ffffffull) | 0x80000000u);   // addr[56:32] | type=2
  v8i g1;
  g1[0] = 1 << 16;                                             // data_size = 2 bytes
  g1[1] = (int)((K & 0xffff) << 16);                           // tensor_dim0[15:0]
  g1[2] = (int)(((K >> 16) & 0xffff) | ((Nout & 0xffff) << 16)); // dim0 hi | dim1 lo
  g1[3] = (int)(((Nout >> 16) & 0xffff) | (64 << 16));         // dim1 hi | tile_dim0=64
  g1[4] = 256;                                                 // tile_dim1=256, tile_dim2=0
  g1[5] = K;                                                   // tensor_dim0_stride lo
  g1[6] = 0;
  g1[7] = 0;
  asm volatile("tensor_load_to_lds %0, %1" :: "s"(g0), "s"(g1) : "memory");
}

// ---------------------------------------------------------------------------
// Segment bookkeeping
// ---------------------------------------------------------------------------
__global__ void zero_counts_kernel(int* counts) { counts[threadIdx.x] = 0; }

__global__ void count_kernel(const int* __restrict__ batch, int* counts, int N) {
  int i = blockIdx.x * blockDim.x + threadIdx.x;
  if (i < N) atomicAdd(&counts[batch[i]], 1);
}

__global__ void scan_kernel(const int* __restrict__ counts, int* starts) {
  int s = 0;
  for (int b = 0; b < NBATCH; ++b) { starts[b] = s; s += counts[b]; }
}

// ---------------------------------------------------------------------------
// f32 -> f16 weight conversion
// ---------------------------------------------------------------------------
__global__ void f32_to_f16_kernel(const float* __restrict__ in, _Float16* __restrict__ out, int n) {
  int i = blockIdx.x * blockDim.x + threadIdx.x;
  if (i < n) out[i] = (_Float16)in[i];
}

// ---------------------------------------------------------------------------
// LayerNorm over D=512, one block (256 threads) per row, f16 output
// ---------------------------------------------------------------------------
__global__ void layernorm_f16_kernel(const float* __restrict__ x,
                                     const float* __restrict__ g,
                                     const float* __restrict__ b,
                                     _Float16* __restrict__ out) {
  __shared__ float s_sum[8];
  __shared__ float s_sq[8];
  const int row = blockIdx.x;
  const float* xr = x + (size_t)row * D_MODEL;
  const int t = threadIdx.x;
  float2 v = ((const float2*)xr)[t];
  float s = v.x + v.y;
  float q = v.x * v.x + v.y * v.y;
  #pragma unroll
  for (int off = 1; off < 32; off <<= 1) {
    s += __shfl_xor(s, off, 32);
    q += __shfl_xor(q, off, 32);
  }
  const int wave = t >> 5, lane = t & 31;
  if (lane == 0) { s_sum[wave] = s; s_sq[wave] = q; }
  __syncthreads();
  s = 0.f; q = 0.f;
  #pragma unroll
  for (int w = 0; w < 8; ++w) { s += s_sum[w]; q += s_sq[w]; }
  const float mean = s * (1.0f / D_MODEL);
  const float var  = q * (1.0f / D_MODEL) - mean * mean;
  const float rs   = rsqrtf(var + 1e-5f);
  const int c0 = 2 * t, c1 = 2 * t + 1;
  out[(size_t)row * D_MODEL + c0] = (_Float16)((v.x - mean) * rs * g[c0] + b[c0]);
  out[(size_t)row * D_MODEL + c1] = (_Float16)((v.y - mean) * rs * g[c1] + b[c1]);
}

// ---------------------------------------------------------------------------
// WMMA GEMM with TDM-staged weights:
//   out[M,Nout] = A[M,K](f16) * W[Nout,K]^T(f16) + bias
// Block = 256 threads (8 waves), block tile 64 rows x 256 cols.
// Wave tile = 32x64 (2 A-frags x 4 B-tiles -> 8 WMMA per 32-k step).
// Weight k-chunks (256x64 halves = 32KB) are DMA'd by the TDM into LDS,
// double-buffered (TENSORcnt), so WMMAs consume B from ds_load while the
// next chunk streams in. A-frags come straight from global, prefetch-ahead.
// Epilogues: 0 = f16 (QKV), 1 = f32 + residual (out-proj/FFN2), 2 = GELU f16.
// ---------------------------------------------------------------------------
__global__ void gemm_wmma_kernel(const _Float16* __restrict__ A,
                                 const _Float16* __restrict__ W,
                                 const float* __restrict__ bias,
                                 const float* __restrict__ residual,
                                 float* __restrict__ outF,
                                 _Float16* __restrict__ outH,
                                 int M, int Nout, int K, int epi) {
  __shared__ _Float16 wlds[2][256][64];   // 2 x 32KB
  const int lane = threadIdx.x & 31;
  const int wv   = threadIdx.x >> 5;
  const int n    = lane & 15;
  const int hsel = lane >> 4;
  const int rowTile  = blockIdx.x * 64 + (wv & 1) * 32;   // 2 row-waves
  const int colWave  = (wv >> 1) * 64;                    // 4 col-waves
  const int colBlock = blockIdx.y * 256;

  v8f zero = {};
  v8f acc[2][4] = {{zero, zero, zero, zero}, {zero, zero, zero, zero}};

  const int nchunks = K >> 6;                             // 64-k chunks
  const _Float16* wbase = W + (size_t)colBlock * K;

  if (wv == 0) {
    unsigned lds0 = (unsigned)(unsigned long long)(&wlds[0][0][0]);
    tdm_load_wtile(wbase, lds0, K, Nout);
  }

  for (int c = 0; c < nchunks; ++c) {
    const int buf = c & 1;
    if (wv == 0) {
      if (c + 1 < nchunks) {
        unsigned ldsn = (unsigned)(unsigned long long)(&wlds[buf ^ 1][0][0]);
        tdm_load_wtile(wbase + (c + 1) * 64, ldsn, K, Nout);
        __builtin_amdgcn_s_wait_tensorcnt(1);   // chunk c has landed
      } else {
        __builtin_amdgcn_s_wait_tensorcnt(0);
      }
    }
    __syncthreads();

    if (c + 1 < nchunks) {                      // prefetch next A chunk
      __builtin_prefetch(A + (size_t)(rowTile + n) * K + (c + 1) * 64, 0, 1);
      __builtin_prefetch(A + (size_t)(rowTile + 16 + n) * K + (c + 1) * 64, 0, 1);
    }

    #pragma unroll
    for (int sub = 0; sub < 64; sub += 32) {
      AFrag af[2];
      #pragma unroll
      for (int a = 0; a < 2; ++a) {
        const _Float16* ap = A + (size_t)(rowTile + a * 16 + n) * K + c * 64 + sub + hsel * 8;
        af[a].h[0] = *(const h8*)ap;
        af[a].h[1] = *(const h8*)(ap + 16);
      }
      #pragma unroll
      for (int t = 0; t < 4; ++t) {
        AFrag bf;
        const _Float16* lp = &wlds[buf][colWave + t * 16 + n][sub + hsel * 8];
        bf.h[0] = *(const h8*)lp;
        bf.h[1] = *(const h8*)(lp + 16);
        acc[0][t] = wmma_f16(af[0].v, bf.v, acc[0][t]);
        acc[1][t] = wmma_f16(af[1].v, bf.v, acc[1][t]);
      }
    }
    __syncthreads();   // protect buf[buf^1]... buf reuse next-next iteration
  }

  #pragma unroll
  for (int a = 0; a < 2; ++a) {
    #pragma unroll
    for (int t = 0; t < 4; ++t) {
      const int col = colBlock + colWave + t * 16 + n;
      const float bv = bias ? bias[col] : 0.0f;
      #pragma unroll
      for (int r = 0; r < 8; ++r) {
        const int row = rowTile + a * 16 + r + 8 * hsel;
        const size_t idx = (size_t)row * Nout + col;
        float v = acc[a][t][r] + bv;
        if (epi == 0) {
          outH[idx] = (_Float16)v;
        } else if (epi == 1) {
          outF[idx] = residual[idx] + v;
        } else {
          outH[idx] = (_Float16)(0.5f * v * (1.0f + erff(v * 0.70710678118f)));
        }
      }
    }
  }
}

// ---------------------------------------------------------------------------
// Ragged flash attention, one block per (graph, head); see round-1 notes.
// ---------------------------------------------------------------------------
__global__ void attn_kernel(const _Float16* __restrict__ qkv,
                            _Float16* __restrict__ ctx,
                            const int* __restrict__ starts,
                            const int* __restrict__ counts) {
  __shared__ float pbuf[8][16][34];
  const int b = blockIdx.x >> 3;
  const int h = blockIdx.x & 7;
  const int start = starts[b];
  const int cnt   = counts[b];
  if (cnt <= 0) return;
  const int lane = threadIdx.x & 31;
  const int wave = threadIdx.x >> 5;
  const int n    = lane & 15;
  const int hsel = lane >> 4;
  const int nqt  = (cnt + 15) >> 4;
  const int nk32 = (cnt + 31) >> 5;
  const float scale = 0.125f; // rsqrt(64)
  v8f zero = {};

  for (int qt = wave; qt < nqt; qt += 8) {
    const int qlocal = qt * 16 + n;
    const int qrow = start + (qlocal < cnt ? qlocal : cnt - 1);
    const _Float16* qp = qkv + (size_t)qrow * TRIPLE + h * DHEAD;
    AFrag aq0, aq1;
    aq0.h[0] = *(const h8*)(qp + hsel * 8);
    aq0.h[1] = *(const h8*)(qp + 16 + hsel * 8);
    aq1.h[0] = *(const h8*)(qp + 32 + hsel * 8);
    aq1.h[1] = *(const h8*)(qp + 48 + hsel * 8);

    float mrow[8], lrow[8];
    v8f acc[4] = {zero, zero, zero, zero};
    #pragma unroll
    for (int r = 0; r < 8; ++r) { mrow[r] = -1.0e30f; lrow[r] = 0.0f; }

    for (int k32 = 0; k32 < nk32; ++k32) {
      v8f s[2];
      #pragma unroll
      for (int st = 0; st < 2; ++st) {
        const int klocal = k32 * 32 + st * 16 + n;
        const int krow = start + (klocal < cnt ? klocal : cnt - 1);
        const _Float16* kp = qkv + (size_t)krow * TRIPLE + D_MODEL + h * DHEAD;
        AFrag bk0, bk1;
        bk0.h[0] = *(const h8*)(kp + hsel * 8);
        bk0.h[1] = *(const h8*)(kp + 16 + hsel * 8);
        bk1.h[0] = *(const h8*)(kp + 32 + hsel * 8);
        bk1.h[1] = *(const h8*)(kp + 48 + hsel * 8);
        v8f sc = zero;
        sc = wmma_f16(aq0.v, bk0.v, sc);
        sc = wmma_f16(aq1.v, bk1.v, sc);
        if (klocal >= cnt) {
          #pragma unroll
          for (int r = 0; r < 8; ++r) sc[r] = NEGBIG;
        } else {
          #pragma unroll
          for (int r = 0; r < 8; ++r) sc[r] *= scale;
        }
        s[st] = sc;
      }
      #pragma unroll
      for (int r = 0; r < 8; ++r) {
        float mx = fmaxf(s[0][r], s[1][r]);
        #pragma unroll
        for (int off = 1; off < 16; off <<= 1)
          mx = fmaxf(mx, __shfl_xor(mx, off, 32));
        const float nm = fmaxf(mrow[r], mx);
        const float corr = __expf(mrow[r] - nm);
        const float p0 = __expf(s[0][r] - nm);
        const float p1 = __expf(s[1][r] - nm);
        float rsum = p0 + p1;
        #pragma unroll
        for (int off = 1; off < 16; off <<= 1)
          rsum += __shfl_xor(rsum, off, 32);
        lrow[r] = lrow[r] * corr + rsum;
        mrow[r] = nm;
        #pragma unroll
        for (int t = 0; t < 4; ++t) acc[t][r] *= corr;
        pbuf[wave][r + 8 * hsel][n]      = p0;
        pbuf[wave][r + 8 * hsel][16 + n] = p1;
      }
      asm volatile("s_wait_dscnt 0" ::: "memory");
      AFrag pa;
      #pragma unroll
      for (int e = 0; e < 8; ++e) {
        pa.e[e]     = (_Float16)pbuf[wave][n][hsel * 8 + e];
        pa.e[e + 8] = (_Float16)pbuf[wave][n][16 + hsel * 8 + e];
      }
      #pragma unroll
      for (int t = 0; t < 4; ++t) {
        AFrag bv;
        #pragma unroll
        for (int e = 0; e < 16; ++e) {
          const int kpat = (e < 8) ? (hsel * 8 + e) : (16 + hsel * 8 + (e - 8));
          const int klocal = k32 * 32 + kpat;
          const int krow = start + (klocal < cnt ? klocal : cnt - 1);
          bv.e[e] = qkv[(size_t)krow * TRIPLE + 2 * D_MODEL + h * DHEAD + t * 16 + n];
        }
        acc[t] = wmma_f16(pa.v, bv.v, acc[t]);
      }
    }
    #pragma unroll
    for (int t = 0; t < 4; ++t) {
      #pragma unroll
      for (int r = 0; r < 8; ++r) {
        const int ql = qt * 16 + r + 8 * hsel;
        if (ql < cnt) {
          const float v = acc[t][r] / fmaxf(lrow[r], 1e-20f);
          ctx[(size_t)(start + ql) * D_MODEL + h * DHEAD + t * 16 + n] = (_Float16)v;
        }
      }
    }
  }
}

// ---------------------------------------------------------------------------
extern "C" void kernel_launch(void* const* d_in, const int* in_sizes, int n_in,
                              void* d_out, int out_size, void* d_ws, size_t ws_size,
                              hipStream_t stream) {
  const float* x          = (const float*)d_in[0];
  const int*   batch      = (const int*)d_in[1];
  const float* in_proj_w  = (const float*)d_in[2];
  const float* in_proj_b  = (const float*)d_in[3];
  const float* out_proj_w = (const float*)d_in[4];
  const float* out_proj_b = (const float*)d_in[5];
  const float* norm1_g    = (const float*)d_in[6];
  const float* norm1_b    = (const float*)d_in[7];
  const float* norm2_g    = (const float*)d_in[8];
  const float* norm2_b    = (const float*)d_in[9];
  const float* ffn_w1     = (const float*)d_in[10];
  const float* ffn_b1     = (const float*)d_in[11];
  const float* ffn_w2     = (const float*)d_in[12];
  const float* ffn_b2     = (const float*)d_in[13];
  float* out = (float*)d_out;
  const int N = in_sizes[1];

  char* ws = (char*)d_ws;
  size_t off = 0;
  auto alloc = [&](size_t bytes) -> char* {
    char* p = ws + off;
    off = (off + bytes + 255) & ~(size_t)255;
    return p;
  };
  int*      counts = (int*)alloc(NBATCH * sizeof(int));
  int*      starts = (int*)alloc(NBATCH * sizeof(int));
  _Float16* h16    = (_Float16*)alloc((size_t)N * D_MODEL * 2);
  _Float16* qkv16  = (_Float16*)alloc((size_t)N * TRIPLE * 2);
  _Float16* ctx16  = (_Float16*)alloc((size_t)N * D_MODEL * 2);
  float*    x2     = (float*)alloc((size_t)N * D_MODEL * 4);
  _Float16* h2     = (_Float16*)alloc((size_t)N * D_MODEL * 2);
  _Float16* a16    = (_Float16*)alloc((size_t)N * D_FF * 2);
  _Float16* wqkv   = (_Float16*)alloc((size_t)TRIPLE * D_MODEL * 2);
  _Float16* wout   = (_Float16*)alloc((size_t)D_MODEL * D_MODEL * 2);
  _Float16* w1     = (_Float16*)alloc((size_t)D_FF * D_MODEL * 2);
  _Float16* w2     = (_Float16*)alloc((size_t)D_MODEL * D_FF * 2);

  zero_counts_kernel<<<1, NBATCH, 0, stream>>>(counts);
  count_kernel<<<(N + 255) / 256, 256, 0, stream>>>(batch, counts, N);
  scan_kernel<<<1, 1, 0, stream>>>(counts, starts);

  f32_to_f16_kernel<<<(TRIPLE * D_MODEL + 255) / 256, 256, 0, stream>>>(in_proj_w, wqkv, TRIPLE * D_MODEL);
  f32_to_f16_kernel<<<(D_MODEL * D_MODEL + 255) / 256, 256, 0, stream>>>(out_proj_w, wout, D_MODEL * D_MODEL);
  f32_to_f16_kernel<<<(D_FF * D_MODEL + 255) / 256, 256, 0, stream>>>(ffn_w1, w1, D_FF * D_MODEL);
  f32_to_f16_kernel<<<(D_MODEL * D_FF + 255) / 256, 256, 0, stream>>>(ffn_w2, w2, D_MODEL * D_FF);

  // LN1 -> QKV
  layernorm_f16_kernel<<<N, 256, 0, stream>>>(x, norm1_g, norm1_b, h16);
  {
    dim3 g(N / 64, TRIPLE / 256);
    gemm_wmma_kernel<<<g, 256, 0, stream>>>(h16, wqkv, in_proj_b, nullptr,
                                            nullptr, qkv16, N, TRIPLE, D_MODEL, 0);
  }
  // ragged flash attention
  attn_kernel<<<NBATCH * NHEAD, 256, 0, stream>>>(qkv16, ctx16, starts, counts);
  // out-proj + residual
  {
    dim3 g(N / 64, D_MODEL / 256);
    gemm_wmma_kernel<<<g, 256, 0, stream>>>(ctx16, wout, out_proj_b, x,
                                            x2, nullptr, N, D_MODEL, D_MODEL, 1);
  }
  // LN2 -> FFN1 (GELU) -> FFN2 + residual
  layernorm_f16_kernel<<<N, 256, 0, stream>>>(x2, norm2_g, norm2_b, h2);
  {
    dim3 g(N / 64, D_FF / 256);
    gemm_wmma_kernel<<<g, 256, 0, stream>>>(h2, w1, ffn_b1, nullptr,
                                            nullptr, a16, N, D_FF, D_MODEL, 2);
  }
  {
    dim3 g(N / 64, D_MODEL / 256);
    gemm_wmma_kernel<<<g, 256, 0, stream>>>(a16, w2, ffn_b2, x2,
                                            out, nullptr, N, D_MODEL, D_FF, 1);
  }
}
